// kl_WeightedKappaLoss_32469952758491
// MI455X (gfx1250) — matrix operand order repaired and checked by reading, not compiled, not run on hardware
//
#include <hip/hip_runtime.h>
#include <hip/hip_bf16.h>

typedef __attribute__((ext_vector_type(16))) _Float16 v16h;
typedef __attribute__((ext_vector_type(8)))  float    v8f;

#define NC 5
#define MAIN_BLOCKS 1024
#define THREADS 256
#define WAVES_PER_BLOCK (THREADS / 32)

// ---------------------------------------------------------------------------
// Kernel 1: confusion-matrix segment-sum via WMMA.
//   conf = OneHot(y_true)^T (16x32 f16, exact) x y_pred (32x16 f16) -> f32 acc.
// Steady-state loop is guard-free and branch-free so loads batch and the wave
// waits once per chunk. Tail rows (n % 32) are handled in the finalize kernel.
// ---------------------------------------------------------------------------
__global__ __launch_bounds__(THREADS)
void kappa_conf_wmma(const float* __restrict__ y_pred,
                     const int*   __restrict__ y_true,
                     float*       __restrict__ partial,   // [gridDim.x * 25]
                     int n) {
  const int tid  = threadIdx.x;
  const int lane = tid & 31;          // wave32
  const int wv   = tid >> 5;
  const int l16  = lane & 15;         // A: row M ; B: col N
  const int hi   = lane >> 4;         // 0: lanes 0-15, 1: lanes 16-31
  const int col  = (l16 < NC) ? l16 : (NC - 1);   // clamped pred column

  const long nfull   = (long)n >> 5;              // full 32-row chunks
  const long wave_id = (long)blockIdx.x * WAVES_PER_BLOCK + wv;
  const long nwaves  = (long)gridDim.x * WAVES_PER_BLOCK;

  v8f c = {};   // 16x16 f32 accumulator; only top-left 5x5 ever nonzero

  for (long chunk = wave_id; chunk < nfull; chunk += nwaves) {
    const long base = chunk << 5;
    const int*   lp = y_true + base;
    const float* pp = y_pred + base * NC;

    // Speculative prefetch of next chunk (global_prefetch_b8; OOB translation
    // failures are silently dropped, so no guard -> no exec divergence).
    __builtin_prefetch(y_pred + (base + (nwaves << 5)) * NC + lane * 4, 0, 1);

    // ---- labels: each half-wave needs K runs [8*hi..8*hi+7] and
    //      [16+8*hi..16+8*hi+7]  ->  4 x global_load_b128 (broadcast). ----
    const int4 q0 = *(const int4*)(lp + (hi << 3));
    const int4 q1 = *(const int4*)(lp + (hi << 3) + 4);
    const int4 q2 = *(const int4*)(lp + 16 + (hi << 3));
    const int4 q3 = *(const int4*)(lp + 16 + (hi << 3) + 4);
    const int lab[16] = {q0.x, q0.y, q0.z, q0.w,  q1.x, q1.y, q1.z, q1.w,
                         q2.x, q2.y, q2.z, q2.w,  q3.x, q3.y, q3.z, q3.w};

    // ---- preds: 16 unconditional b32 loads (rows h+(h&8)+8*hi, clamped col);
    //      no branches, so all 16 issue before a single wait. ----
    float pv[16];
#pragma unroll
    for (int h = 0; h < 16; ++h)
      pv[h] = pp[((hi << 3) + h + (h & 8)) * NC + col];

    // ---- build fragments: pure VALU (cndmask / cvt), no control flow ----
    v16h a, b;
#pragma unroll
    for (int h = 0; h < 16; ++h) {
      // A layout (16-bit A 16x32): lane M=l16, K = h + (h&8) + 8*hi.
      a[h] = (lab[h] == l16) ? (_Float16)1.0f : (_Float16)0.0f;
      // B layout (16-bit B 32x16): lane N=l16, K = h + (h&8) + 8*hi; cols >=5 -> 0.
      b[h] = (l16 < NC) ? (_Float16)pv[h] : (_Float16)0.0f;
    }

    c = __builtin_amdgcn_wmma_f32_16x16x32_f16(
        /*neg_a=*/false, a, /*neg_b=*/false, b,
        /*c_mod=*/(short)0, c, /*reuse_a=*/false, /*reuse_b=*/false);
  }

  // ---- Block reduction: 8 wave fragments -> one 5x5 partial.
  // C/D layout: VGPR r, lane l (l<16)  <->  conf[M=r][N=l].
  __shared__ float red[WAVES_PER_BLOCK][32][8];   // 8 KB
#pragma unroll
  for (int r = 0; r < 8; ++r) red[wv][lane][r] = c[r];
  __syncthreads();

  if (tid < NC * NC) {
    const int r = tid / NC;   // M (true class)
    const int q = tid % NC;   // N (pred column)
    float s = 0.0f;
#pragma unroll
    for (int w = 0; w < WAVES_PER_BLOCK; ++w) s += red[w][q][r];
    partial[(long)blockIdx.x * (NC * NC) + tid] = s;
  }
}

// ---------------------------------------------------------------------------
// Kernel 2: reduce per-block partials -> conf[5][5], add tail rows, then the
// scalar kappa math. Launched with 800 threads (25 elements x 32 reducers).
// ---------------------------------------------------------------------------
__global__ __launch_bounds__(800)
void kappa_finalize(const float* __restrict__ partial, int nblocks,
                    const float* __restrict__ y_pred,
                    const int*   __restrict__ y_true, int n,
                    float* __restrict__ out) {
  __shared__ float sred[NC * NC][32];
  __shared__ float conf_s[NC * NC];
  const int t = threadIdx.x;

  {
    const int e = t >> 5;     // conf element 0..24
    const int i = t & 31;
    float s = 0.0f;
    for (int b = i; b < nblocks; b += 32)
      s += partial[(long)b * (NC * NC) + e];
    sred[e][i] = s;
  }
  __syncthreads();

  if (t < NC * NC) {
    float s = 0.0f;
#pragma unroll
    for (int i = 0; i < 32; ++i) s += sred[t][i];
    conf_s[t] = s;
  }
  __syncthreads();

  if (t == 0) {
    float conf[NC][NC];
#pragma unroll
    for (int i = 0; i < NC; ++i)
#pragma unroll
      for (int j = 0; j < NC; ++j) conf[i][j] = conf_s[i * NC + j];

    // Tail rows not covered by 32-row WMMA chunks (n % 32 rows, <= 31).
    const int tail_start = (int)(((long)n >> 5) << 5);
    for (int r = tail_start; r < n; ++r) {
      const int labr = y_true[r];
#pragma unroll
      for (int j = 0; j < NC; ++j) conf[labr][j] += y_pred[(long)r * NC + j];
    }

    float row[NC], col[NC], total = 0.0f;
#pragma unroll
    for (int i = 0; i < NC; ++i) { row[i] = 0.0f; col[i] = 0.0f; }
#pragma unroll
    for (int i = 0; i < NC; ++i)
#pragma unroll
      for (int j = 0; j < NC; ++j) {
        const float v = conf[i][j];
        row[i] += v;
        col[j] += v;
        total  += v;
      }

    // weight_matrix: |i-j|^0.5 / (NC-1)^0.5 = sqrt(|i-j|) / 2
    float num = 0.0f, den = 0.0f;
#pragma unroll
    for (int i = 0; i < NC; ++i)
#pragma unroll
      for (int j = 0; j < NC; ++j) {
        const float w = sqrtf(fabsf((float)(i - j))) * 0.5f;
        num += w * conf[i][j];
        den += w * row[i] * col[j];
      }
    den /= total;
    const float kappa = 1.0f - num / den;

    float pen = 0.0f;
#pragma unroll
    for (int i = 0; i < NC; ++i)
      pen += row[i] * (logf(row[i] + 1e-8f) - logf(col[i] + 1e-8f));
    pen /= total;

    out[0] = -logf((kappa + 1.0f) * 0.5f) + pen;
  }
}

// ---------------------------------------------------------------------------
extern "C" void kernel_launch(void* const* d_in, const int* in_sizes, int n_in,
                              void* d_out, int out_size, void* d_ws, size_t ws_size,
                              hipStream_t stream) {
  const float* y_pred = (const float*)d_in[0];   // [BATCH, 5] f32
  const int*   y_true = (const int*)d_in[1];     // [BATCH]    i32
  const int    n      = in_sizes[1];

  float* partial = (float*)d_ws;                 // MAIN_BLOCKS * 25 floats
  float* out     = (float*)d_out;                // single f32 loss

  kappa_conf_wmma<<<MAIN_BLOCKS, THREADS, 0, stream>>>(y_pred, y_true, partial, n);
  kappa_finalize<<<1, 800, 0, stream>>>(partial, MAIN_BLOCKS, y_pred, y_true, n, out);
}
